// SpaceTimeSelfAttention_2525440770136
// MI455X (gfx1250) — compile-verified, matrix operand order checked by reading, not run on hardware
//
#include <hip/hip_runtime.h>

typedef __attribute__((ext_vector_type(16))) __bf16 v16bf;
typedef __attribute__((ext_vector_type(8)))  __bf16 v8bf;
typedef __attribute__((ext_vector_type(8)))  float  v8f;
typedef __attribute__((ext_vector_type(4)))  float  v4f;
typedef __attribute__((ext_vector_type(4)))  int    v4i;

#define NHEADS 12
#define HDIM   768
#define DHEAD  64
#define SEQ    1024
#define BATCH  4
#define NROWS  (BATCH*SEQ)                            // 4096
#define PROJ_ELEMS ((size_t)BATCH*NHEADS*SEQ*DHEAD)   // 3,145,728 per matrix
#define NROWTOT (BATCH*NHEADS*SEQ)                    // 49152
#define XELEMS ((size_t)NROWS*HDIM)                   // 3,145,728
#define WELEMS ((size_t)HDIM*HDIM)                    // 589,824

#ifndef __has_builtin
#define __has_builtin(x) 0
#endif

// -------- CDNA5 async global->LDS copy (ASYNCcnt path), builtin or asm ------
// Probe round 2 showed the builtin exists with (AS1 int4* src, AS3 int4* dst,
// imm offset, imm cpol) — cast through uintptr to the exact pointer types.
#if __has_builtin(__builtin_amdgcn_global_load_async_to_lds_b128)
#define ASYNC_CP16(gp, lp)                                                     \
    __builtin_amdgcn_global_load_async_to_lds_b128(                            \
        (__attribute__((address_space(1))) v4i*)(size_t)(const void*)(gp),     \
        (__attribute__((address_space(3))) v4i*)(unsigned)(size_t)(const void*)(lp), \
        0, 0)
#else
#define ASYNC_CP16(gp, lp)                                                     \
    asm volatile("global_load_async_to_lds_b128 %0, %1, off"                   \
                 :: "v"((unsigned)(size_t)(const void*)(lp)),                  \
                    "v"((unsigned long long)(size_t)(const void*)(gp))         \
                 : "memory")
#endif

#if __has_builtin(__builtin_amdgcn_s_wait_asynccnt)
#define WAIT_ASYNC0() __builtin_amdgcn_s_wait_asynccnt(0)
#else
#define WAIT_ASYNC0() asm volatile("s_wait_asynccnt 0x0" ::: "memory")
#endif

// ---------------------------------------------------------------------------
__device__ __forceinline__ v16bf cat16(v8bf lo, v8bf hi) {
    return __builtin_shufflevector(lo, hi, 0,1,2,3,4,5,6,7,8,9,10,11,12,13,14,15);
}

// A/B fragment (16x32 bf16): per lane, 8 values at K=base..base+7 and 8 at
// K=base+16..base+23. Caller pre-offsets p by (lane>>4)*8.
__device__ __forceinline__ v16bf frag_from_bf16(const __bf16* p) {
    v8bf lo = *(const v8bf*)(p);
    v8bf hi = *(const v8bf*)(p + 16);
    return cat16(lo, hi);
}

// ---------------------------------------------------------------------------
// Kernel 0: f32 -> bf16 streaming conversion (8 elements / thread)
// ---------------------------------------------------------------------------
__global__ __launch_bounds__(256)
void cvt_kernel(const float* __restrict__ src, __bf16* __restrict__ dst, int n8)
{
    const int t = blockIdx.x * blockDim.x + threadIdx.x;
    if (t >= n8) return;
    const v4f a = *(const v4f*)(src + (size_t)t * 8);
    const v4f b = *(const v4f*)(src + (size_t)t * 8 + 4);
    v8bf o;
#pragma unroll
    for (int i = 0; i < 4; ++i) { o[i] = (__bf16)a[i]; o[4 + i] = (__bf16)b[i]; }
    *(v8bf*)(dst + (size_t)t * 8) = o;
}

// ---------------------------------------------------------------------------
// Kernel 1: projection GEMM, bf16 in / bf16 out. One wave computes a 32x64
// tile of X @ W^T + b, stored in [B,NH,S,DH] layout.
// ---------------------------------------------------------------------------
__global__ __launch_bounds__(32)
void proj_kernel(const __bf16* __restrict__ Xb, const __bf16* __restrict__ Wb,
                 const float* __restrict__ bias, __bf16* __restrict__ dst)
{
    const int lane = threadIdx.x & 31;
    const int hsel = lane >> 4;
    const int lm   = lane & 15;
    const int m0   = blockIdx.x * 32;
    const int n0   = blockIdx.y * 64;

    const __bf16* a0p = Xb + (size_t)(m0 + lm) * HDIM + hsel * 8;
    const __bf16* a1p = Xb + (size_t)(m0 + 16 + lm) * HDIM + hsel * 8;
    const __bf16* b0p = Wb + (size_t)(n0 +  0 + lm) * HDIM + hsel * 8;
    const __bf16* b1p = Wb + (size_t)(n0 + 16 + lm) * HDIM + hsel * 8;
    const __bf16* b2p = Wb + (size_t)(n0 + 32 + lm) * HDIM + hsel * 8;
    const __bf16* b3p = Wb + (size_t)(n0 + 48 + lm) * HDIM + hsel * 8;

    v8f acc[2][4];
#pragma unroll
    for (int mb = 0; mb < 2; ++mb)
#pragma unroll
        for (int nb = 0; nb < 4; ++nb) acc[mb][nb] = v8f{};

#pragma unroll 2
    for (int k0 = 0; k0 < HDIM; k0 += 32) {
        v16bf a0 = frag_from_bf16(a0p + k0);
        v16bf a1 = frag_from_bf16(a1p + k0);
        v16bf b0 = frag_from_bf16(b0p + k0);
        v16bf b1 = frag_from_bf16(b1p + k0);
        v16bf b2 = frag_from_bf16(b2p + k0);
        v16bf b3 = frag_from_bf16(b3p + k0);
        acc[0][0] = __builtin_amdgcn_wmma_f32_16x16x32_bf16(false, a0, false, b0, (short)0, acc[0][0], false, false);
        acc[0][1] = __builtin_amdgcn_wmma_f32_16x16x32_bf16(false, a0, false, b1, (short)0, acc[0][1], false, false);
        acc[0][2] = __builtin_amdgcn_wmma_f32_16x16x32_bf16(false, a0, false, b2, (short)0, acc[0][2], false, false);
        acc[0][3] = __builtin_amdgcn_wmma_f32_16x16x32_bf16(false, a0, false, b3, (short)0, acc[0][3], false, false);
        acc[1][0] = __builtin_amdgcn_wmma_f32_16x16x32_bf16(false, a1, false, b0, (short)0, acc[1][0], false, false);
        acc[1][1] = __builtin_amdgcn_wmma_f32_16x16x32_bf16(false, a1, false, b1, (short)0, acc[1][1], false, false);
        acc[1][2] = __builtin_amdgcn_wmma_f32_16x16x32_bf16(false, a1, false, b2, (short)0, acc[1][2], false, false);
        acc[1][3] = __builtin_amdgcn_wmma_f32_16x16x32_bf16(false, a1, false, b3, (short)0, acc[1][3], false, false);
    }

#pragma unroll
    for (int nb = 0; nb < 4; ++nb) {
        const int col = n0 + nb * 16 + lm;
        const float bn = bias[col];
        const int h = col >> 6, dh = col & 63;
#pragma unroll
        for (int mb = 0; mb < 2; ++mb) {
#pragma unroll
            for (int r = 0; r < 8; ++r) {
                const int m = m0 + mb * 16 + r + 8 * hsel;   // C-layout rows
                const int bI = m >> 10, s = m & 1023;
                dst[(((size_t)(bI * NHEADS + h) * SEQ + s) << 6) + dh] =
                    (__bf16)(acc[mb][nb][r] + bn);
            }
        }
    }
}

// ---------------------------------------------------------------------------
// Kernel 2: per-row reciprocal norms 1/(||x||+eps) for T then Sp.
// ---------------------------------------------------------------------------
__global__ __launch_bounds__(256)
void norm_kernel(const __bf16* __restrict__ proj_base, float* __restrict__ rn)
{
    const int tid = blockIdx.x * blockDim.x + threadIdx.x;
    if (tid >= 2 * NROWTOT) return;
    const int which = tid / NROWTOT;
    const int row   = tid - which * NROWTOT;
    const __bf16* p = proj_base + (size_t)(3 + which) * PROJ_ELEMS + (size_t)row * DHEAD;
    float s = 0.f;
#pragma unroll
    for (int i = 0; i < DHEAD; ++i) { float v = (float)p[i]; s += v * v; }
    rn[tid] = 1.0f / (sqrtf(s) + 1e-6f);
}

// ---------------------------------------------------------------------------
// Kernel 3: fused flash attention with triple-score product.
// Block = 4 waves, each owning 16 q rows. k-blocks of 32 columns.
// K/T/Sp tiles staged via async global->LDS (single fetch per block);
// V staged transposed; P re-laid out through wave-private LDS.
// ---------------------------------------------------------------------------
__global__ __launch_bounds__(128)
void attn_kernel(const __bf16* __restrict__ proj_base, const float* __restrict__ rn,
                 const float* __restrict__ mask, float* __restrict__ out)
{
    const __bf16* Qp = proj_base;
    const __bf16* Kp = proj_base + PROJ_ELEMS;
    const __bf16* Vp = proj_base + 2 * PROJ_ELEMS;
    const __bf16* Tp = proj_base + 3 * PROJ_ELEMS;
    const __bf16* Sp = proj_base + 4 * PROJ_ELEMS;
    const float* rnT = rn;
    const float* rnS = rn + NROWTOT;

    __shared__ __bf16 Kt[32][72];     // row stride 144B: 16B aligned, conflict-light
    __shared__ __bf16 Tt[32][72];
    __shared__ __bf16 St[32][72];
    __shared__ __bf16 Vt[64][40];     // V tile transposed: Vt[dh][k]
    __shared__ __bf16 pP[4][16][40];  // per-wave P staging

    const int bh   = blockIdx.x;
    const int b    = bh / NHEADS, h = bh % NHEADS;
    const int wave = threadIdx.x >> 5;
    const int lane = threadIdx.x & 31;
    const int hsel = lane >> 4;
    const int lm   = lane & 15;
    const int q0   = blockIdx.y * 64 + wave * 16;

    const __bf16* qrow = Qp + ((size_t)(bh * SEQ + q0 + lm) << 6) + hsel * 8;
    const __bf16* trow = Tp + ((size_t)(bh * SEQ + q0 + lm) << 6) + hsel * 8;
    const __bf16* srow = Sp + ((size_t)(bh * SEQ + q0 + lm) << 6) + hsel * 8;
    v16bf aQ[2], aT[2], aS[2];
#pragma unroll
    for (int c = 0; c < 2; ++c) {
        aQ[c] = frag_from_bf16(qrow + c * 32);
        aT[c] = frag_from_bf16(trow + c * 32);
        aS[c] = frag_from_bf16(srow + c * 32);
    }

    float rnTq[8], rnSq[8], mI[8], lI[8];
    v8f O[4] = {v8f{}, v8f{}, v8f{}, v8f{}};
#pragma unroll
    for (int r = 0; r < 8; ++r) {
        const int qi = bh * SEQ + q0 + r + 8 * hsel;
        rnTq[r] = rnT[qi];
        rnSq[r] = rnS[qi];
        mI[r] = -3.0e38f;
        lI[r] = 0.f;
    }

    for (int k0 = 0; k0 < SEQ; k0 += 32) {
        __syncthreads();   // previous iteration's LDS readers are done

        // --- async-stage K/T/Sp tiles (3 x 32 rows x 8 x 16B chunks) ---
        for (int c = threadIdx.x; c < 768; c += 128) {
            const int t  = c >> 8;
            const int e  = c & 255;
            const int kk = e >> 3, sg = e & 7;
            const size_t goff = ((size_t)(bh * SEQ + k0 + kk) << 6) + sg * 8;
            if (t == 0)      ASYNC_CP16(Kp + goff, &Kt[kk][sg * 8]);
            else if (t == 1) ASYNC_CP16(Tp + goff, &Tt[kk][sg * 8]);
            else             ASYNC_CP16(Sp + goff, &St[kk][sg * 8]);
        }
        // --- stage V tile transposed (b128 global reads, b16 LDS scatter) ---
        for (int c = threadIdx.x; c < 256; c += 128) {
            const int kk = c >> 3, sg = c & 7;
            v8bf vv = *(const v8bf*)(Vp + ((size_t)(bh * SEQ + k0 + kk) << 6) + sg * 8);
#pragma unroll
            for (int i = 0; i < 8; ++i) Vt[sg * 8 + i][kk] = vv[i];
        }
        WAIT_ASYNC0();
        __syncthreads();

        // --- scores: 16 q rows x 32 k cols, three matrices fused ---
        v8f sb[2];
#pragma unroll
        for (int j = 0; j < 2; ++j) {
            const int rr = j * 16 + lm;
            v8f accB = {}, accT = {}, accS = {};
#pragma unroll
            for (int c = 0; c < 2; ++c) {
                const int off = c * 32 + hsel * 8;
                v16bf bK = cat16(*(const v8bf*)&Kt[rr][off], *(const v8bf*)&Kt[rr][off + 16]);
                v16bf bT = cat16(*(const v8bf*)&Tt[rr][off], *(const v8bf*)&Tt[rr][off + 16]);
                v16bf bS = cat16(*(const v8bf*)&St[rr][off], *(const v8bf*)&St[rr][off + 16]);
                accB = __builtin_amdgcn_wmma_f32_16x16x32_bf16(false, aQ[c], false, bK, (short)0, accB, false, false);
                accT = __builtin_amdgcn_wmma_f32_16x16x32_bf16(false, aT[c], false, bT, (short)0, accT, false, false);
                accS = __builtin_amdgcn_wmma_f32_16x16x32_bf16(false, aS[c], false, bS, (short)0, accS, false, false);
            }
            const float mval = mask[b * SEQ + k0 + j * 16 + lm];
#pragma unroll
            for (int r = 0; r < 8; ++r)
                sb[j][r] = accB[r] * (accT[r] * rnTq[r]) * (accS[r] * rnSq[r]) * 0.125f + mval;
        }

        // --- online softmax across 32 columns; row = (r, lane-half) ---
#pragma unroll
        for (int r = 0; r < 8; ++r) {
            float v = fmaxf(sb[0][r], sb[1][r]);
#pragma unroll
            for (int off = 1; off < 16; off <<= 1) v = fmaxf(v, __shfl_xor(v, off, 16));
            const float M2 = fmaxf(mI[r], v);
            const float al = __expf(mI[r] - M2);
            const float p0 = __expf(sb[0][r] - M2);
            const float p1 = __expf(sb[1][r] - M2);
            float rs = p0 + p1;
#pragma unroll
            for (int off = 1; off < 16; off <<= 1) rs += __shfl_xor(rs, off, 16);
            mI[r] = M2;
            lI[r] = lI[r] * al + rs;
            O[0][r] *= al; O[1][r] *= al; O[2][r] *= al; O[3][r] *= al;
            const int m = r + 8 * hsel;
            pP[wave][m][lm]      = (__bf16)p0;
            pP[wave][m][16 + lm] = (__bf16)p1;
        }

        // --- P A-fragment from wave-private LDS (DS ops in-order per wave) ---
        v16bf aP = cat16(*(const v8bf*)&pP[wave][lm][hsel * 8],
                         *(const v8bf*)&pP[wave][lm][16 + hsel * 8]);
#pragma unroll
        for (int nb = 0; nb < 4; ++nb) {
            v16bf bV = cat16(*(const v8bf*)&Vt[nb * 16 + lm][hsel * 8],
                             *(const v8bf*)&Vt[nb * 16 + lm][16 + hsel * 8]);
            O[nb] = __builtin_amdgcn_wmma_f32_16x16x32_bf16(false, aP, false, bV, (short)0, O[nb], false, false);
        }
    }

#pragma unroll
    for (int r = 0; r < 8; ++r) {
        const float inv = 1.0f / lI[r];
        const int s = q0 + r + 8 * hsel;
        float* op = out + (size_t)(b * SEQ + s) * HDIM + h * DHEAD + lm;
        op[0]  = O[0][r] * inv;
        op[16] = O[1][r] * inv;
        op[32] = O[2][r] * inv;
        op[48] = O[3][r] * inv;
    }
}

// ---------------------------------------------------------------------------
extern "C" void kernel_launch(void* const* d_in, const int* in_sizes, int n_in,
                              void* d_out, int out_size, void* d_ws, size_t ws_size,
                              hipStream_t stream)
{
    const float* hidden = (const float*)d_in[0];
    const float* timee  = (const float*)d_in[1];
    const float* space  = (const float*)d_in[2];
    const float* mask   = (const float*)d_in[3];
    const float* W[5] = {(const float*)d_in[4], (const float*)d_in[6],
                         (const float*)d_in[8], (const float*)d_in[10],
                         (const float*)d_in[12]};
    const float* bias[5] = {(const float*)d_in[5], (const float*)d_in[7],
                            (const float*)d_in[9], (const float*)d_in[11],
                            (const float*)d_in[13]};
    float* out = (float*)d_out;

    // workspace layout
    __bf16* proj = (__bf16*)d_ws;
    __bf16* Xbf  = proj + 5 * PROJ_ELEMS;              // 3 converted activations
    __bf16* Wbf  = Xbf + 3 * XELEMS;                   // 5 converted weights
    float*  rn   = (float*)(Wbf + 5 * WELEMS);         // 2*NROWTOT f32 (4B aligned)

    // 0) f32 -> bf16 conversions
    const float* xsrc[3] = {hidden, timee, space};
    for (int i = 0; i < 3; ++i)
        cvt_kernel<<<dim3((int)(XELEMS / 8 + 255) / 256), 256, 0, stream>>>(
            xsrc[i], Xbf + (size_t)i * XELEMS, (int)(XELEMS / 8));
    for (int i = 0; i < 5; ++i)
        cvt_kernel<<<dim3((int)(WELEMS / 8 + 255) / 256), 256, 0, stream>>>(
            W[i], Wbf + (size_t)i * WELEMS, (int)(WELEMS / 8));

    // 1) five projection GEMMs (q,k,v from hidden; T from time; Sp from space)
    const int xsel[5] = {0, 0, 0, 1, 2};
    for (int w = 0; w < 5; ++w)
        proj_kernel<<<dim3(NROWS / 32, HDIM / 64), 32, 0, stream>>>(
            Xbf + (size_t)xsel[w] * XELEMS, Wbf + (size_t)w * WELEMS,
            bias[w], proj + (size_t)w * PROJ_ELEMS);

    // 2) reciprocal norms for T and Sp
    norm_kernel<<<dim3((2 * NROWTOT + 255) / 256), 256, 0, stream>>>(proj, rn);

    // 3) fused attention
    attn_kernel<<<dim3(BATCH * NHEADS, SEQ / 64), 128, 0, stream>>>(proj, rn, mask, out);
}
